// Encoder_43559558316276
// MI455X (gfx1250) — matrix operand order repaired and checked by reading, not compiled, not run on hardware
//
#include <hip/hip_runtime.h>
#include <hip/hip_bf16.h>
#include <cstdint>
#include <cstddef>

typedef __attribute__((ext_vector_type(16))) __bf16 v16bf;
typedef __attribute__((ext_vector_type(8)))  float  v8f;
typedef __attribute__((ext_vector_type(4)))  float  v4f;

// ---------------------------------------------------------------------------
// Kernel 1: per-node edge-typed gather/scatter into (N, Kpad) bf16 agg matrix.
// One wave per node; node's 7 edges processed serially -> no atomics needed.
// agg row = [slot0: x[col]+onehot | slot1 | ... | slot6 | zero pad to Kpad]
// ---------------------------------------------------------------------------
__global__ __launch_bounds__(256) void gather_agg_kernel(
    const float* __restrict__ x, const int* __restrict__ col,
    const int* __restrict__ etype, const int* __restrict__ ntype,
    int C, int nt, int Kpad, __bf16* __restrict__ agg, int nNodes) {
  extern __shared__ float lds[];
  const int lane = threadIdx.x & 31;
  const int wave = threadIdx.x >> 5;
  const int node = blockIdx.x * (blockDim.x >> 5) + wave;
  if (node >= nNodes) return;
  float* row = lds + (size_t)wave * Kpad;
  for (int k = lane; k < Kpad; k += 32) row[k] = 0.0f;
  const int Ca = C + nt;
  for (int k = 0; k < 7; ++k) {
    const int e = node * 7 + k;
    const int c = col[e];
    const int t = etype[e];
    const int base = t * Ca;
    const float* xr = x + (size_t)c * C;
    for (int ch = lane; ch < C; ch += 32) row[base + ch] += xr[ch];
    if (lane == 0) row[base + C + ntype[c]] += 1.0f;
  }
  __bf16* out = agg + (size_t)node * Kpad;
  for (int k = lane; k < Kpad; k += 32) out[k] = (__bf16)row[k];
}

// ---------------------------------------------------------------------------
// Kernel 2: pack f32 weight (Kreal, Cout) into bf16 WMMA-B fragment layout.
// Index: (((kt*ntiles + ntile)*32 + lane)*16 + t), K = kt*32 + (t<8?0:16) + (lane<16?0:8) + (t&7)
// ---------------------------------------------------------------------------
__global__ __launch_bounds__(256) void pack_w_kernel(
    const float* __restrict__ w, int Kreal, int Kpad, int Cout,
    __bf16* __restrict__ wpack) {
  const int ntiles = Cout >> 4;
  const int total = (Kpad >> 5) * ntiles * 512;
  const int idx = blockIdx.x * blockDim.x + threadIdx.x;
  if (idx >= total) return;
  const int t = idx & 15;
  const int lane = (idx >> 4) & 31;
  const int tile = idx >> 9;
  const int ntile = tile % ntiles;
  const int kt = tile / ntiles;
  const int krow = kt * 32 + ((t >> 3) << 4) + ((lane >> 4) << 3) + (t & 7);
  const int ncol = ntile * 16 + (lane & 15);
  float v = (krow < Kreal) ? w[(size_t)krow * Cout + ncol] : 0.0f;
  wpack[idx] = (__bf16)v;
}

// ---------------------------------------------------------------------------
// Kernel 3: bf16 WMMA GEMM  out(M,Cout) = agg(M,Kpad) @ Wpack.
// Block = 128 threads = 4 waves; each wave owns one 16x16 output tile,
// block covers 64 rows x 16 cols. A loaded per ISA 16-bit A layout, B from
// pre-packed fragments, accumulate f32 via v_wmma_f32_16x16x32_bf16.
// ---------------------------------------------------------------------------
__global__ __launch_bounds__(128) void gemm_bf16_kernel(
    const __bf16* __restrict__ agg, const __bf16* __restrict__ wpack,
    float* __restrict__ out, int M, int K, int Cout) {
  const int lane = threadIdx.x & 31;
  const int wave = threadIdx.x >> 5;
  const int m0 = blockIdx.x * 64 + wave * 16;
  if (m0 >= M) return;
  const int ntile = blockIdx.y;
  const int ntiles = Cout >> 4;
  const int row = m0 + (lane & 15);
  const __bf16* arow = agg + (size_t)row * K;
  const int kb = (lane < 16) ? 0 : 8;  // half-index base within 32-K tile
  v8f acc = {};
  const int ktiles = K >> 5;
  for (int kt = 0; kt < ktiles; ++kt) {
    union { v16bf v; v4f q[2]; } ua;
    ua.q[0] = *(const v4f*)(arow + kt * 32 + kb);
    ua.q[1] = *(const v4f*)(arow + kt * 32 + kb + 16);
    v16bf bfrag = *(const v16bf*)(wpack +
        (((size_t)(kt * ntiles + ntile)) * 32 + lane) * 16);
    acc = __builtin_amdgcn_wmma_f32_16x16x32_bf16(
        false, ua.v, false, bfrag, (short)0, acc, false, false);
  }
  const int colo = ntile * 16 + (lane & 15);
  const int rb = (lane >> 4) * 8;  // lanes>=16 hold rows M=8..15
#pragma unroll
  for (int r = 0; r < 8; ++r)
    out[(size_t)(m0 + rb + r) * Cout + colo] = acc[r];
}

// ---------------------------------------------------------------------------
// Kernel 4: GroupNorm + (optional pre-GELU residual) + GELU(tanh) +
// (optional post-GELU skip add). shift=3 implements repeat-8 upsample
// (GN of repeated rows == repeated GN of rows).
// One thread per (node, group); group size C/grp is 4 or 8.
// ---------------------------------------------------------------------------
__global__ __launch_bounds__(256) void gn_act_kernel(
    const float* __restrict__ h, const float* __restrict__ g,
    const float* __restrict__ bb, const float* __restrict__ resPre,
    const float* __restrict__ addPost, float* __restrict__ out,
    int nOut, int C, int grp, int shift) {
  const int gid = blockIdx.x * blockDim.x + threadIdx.x;
  if (gid >= nOut * grp) return;
  const int node = gid / grp;
  const int gi = gid % grp;
  const int gsize = C / grp;  // 4 or 8
  const int cbase = gi * gsize;
  const float* hr = h + (size_t)(node >> shift) * C + cbase;
  float vals[8];
  float mu = 0.0f;
  for (int j = 0; j < gsize; ++j) { vals[j] = hr[j]; mu += vals[j]; }
  mu /= (float)gsize;
  float var = 0.0f;
  for (int j = 0; j < gsize; ++j) { float d = vals[j] - mu; var += d * d; }
  var /= (float)gsize;
  const float inv = rsqrtf(var + 1e-5f);
  for (int j = 0; j < gsize; ++j) {
    const int c = cbase + j;
    float y = (vals[j] - mu) * inv * g[c] + bb[c];
    if (resPre) y += resPre[(size_t)node * C + c];
    // jax.nn.gelu default: tanh approximation
    float t = tanhf(0.7978845608028654f * (y + 0.044715f * y * y * y));
    float z = 0.5f * y * (1.0f + t);
    if (addPost) z += addPost[(size_t)node * C + c];
    out[(size_t)node * C + c] = z;
  }
}

// ---------------------------------------------------------------------------
// Kernel 5: mean-pool over P children (P=8 for downsample, P=1 = pure
// f32 -> bf16 convert for upsample 1x1) into a GEMM-ready bf16 matrix.
// ---------------------------------------------------------------------------
__global__ __launch_bounds__(256) void pool_bf16_kernel(
    const float* __restrict__ x, __bf16* __restrict__ out,
    int nOut, int C, int P) {
  const int gid = blockIdx.x * blockDim.x + threadIdx.x;
  if (gid >= nOut * C) return;
  const int n = gid / C;
  const int c = gid % C;
  float s = 0.0f;
  for (int p = 0; p < P; ++p) s += x[(size_t)(n * P + p) * C + c];
  out[(size_t)n * C + c] = (__bf16)(s / (float)P);
}

// ---------------------------------------------------------------------------
// Host orchestration
// ---------------------------------------------------------------------------
static const int kN[5]  = {262144, 32768, 4096, 512, 64};
static const int kNT[5] = {7, 6, 5, 4, 3};

// Params pytree leaves in JAX flatten order (dict keys sorted, {b,g,w}).
static const int kLeafSize[99] = {
  32, 32, 2912,                                   // conv1
  64, 64, 2048,                                   // down0
  64, 64, 4096,                                   // down1
  32, 32, 8736,                                   // enc0
  64, 64, 31360,                                  // enc1
  256, 256, 464128, 256, 256, 464128,             // net_dec0_0 (c1,c2)
  256, 256, 464128, 256, 256, 464128,             // net_dec0_1
  128, 128, 118272, 128, 128, 118272,             // net_dec1_0
  128, 128, 118272, 128, 128, 118272,             // net_dec1_1
  64, 64, 30912, 64, 64, 30912,                   // net_dec2_0
  64, 64, 30912, 64, 64, 30912,                   // net_dec2_1
  128, 128, 8192,                                 // net_down0
  256, 256, 32768,                                // net_down1
  64, 64, 30912, 64, 64, 30912,                   // net_enc0_0
  64, 64, 30912, 64, 64, 30912,                   // net_enc0_1
  128, 128, 118272, 128, 128, 118272,             // net_enc1_0
  128, 128, 118272, 128, 128, 118272,             // net_enc1_1
  256, 256, 464128, 256, 256, 464128,             // net_enc2_0
  256, 256, 464128, 256, 256, 464128,             // net_enc2_1
  128, 128, 32768,                                // net_up0
  64, 64, 8192                                    // net_up1
};

static inline int pad32(int k) { return (k + 31) & ~31; }

extern "C" void kernel_launch(void* const* d_in, const int* in_sizes, int n_in,
                              void* d_out, int out_size, void* d_ws, size_t ws_size,
                              hipStream_t stream) {
  (void)in_sizes; (void)out_size; (void)ws_size;
  const float* data = (const float*)d_in[0];

  // Params access: either 99 separate leaf pointers, or one concat buffer.
  const bool split = (n_in > 20);
  size_t offs[100];
  offs[0] = 0;
  for (int i = 0; i < 99; ++i) offs[i + 1] = offs[i] + (size_t)kLeafSize[i];
  const float* pbase = (const float*)d_in[1];
  auto leaf = [&](int k) -> const float* {
    return split ? (const float*)d_in[1 + k] : pbase + offs[k];
  };
  const int gbase = split ? 100 : 2;
  const int *col[5], *ety[5], *nty[5];
  for (int i = 0; i < 5; ++i) {
    col[i] = (const int*)d_in[gbase + 3 * i + 0];
    ety[i] = (const int*)d_in[gbase + 3 * i + 1];
    nty[i] = (const int*)d_in[gbase + 3 * i + 2];
  }

  // Workspace carving (256-byte aligned regions).
  size_t off = 0;
  auto carve = [&](size_t bytes) -> void* {
    void* p = (char*)d_ws + off;
    off += (bytes + 255) & ~(size_t)255;
    return p;
  };
  __bf16* agg    = (__bf16*)carve((size_t)262144 * 288 * 2);  // max agg
  __bf16* wpack  = (__bf16*)carve((size_t)57 * 16 * 512 * 2); // max packed W
  __bf16* pooled = (__bf16*)carve((size_t)32768 * 64 * 2);
  float* bufH = (float*)carve((size_t)262144 * 32 * 4);
  float* bufT = (float*)carve((size_t)262144 * 32 * 4);
  float* bufX = (float*)carve((size_t)262144 * 32 * 4);
  float* X2 = (float*)carve((size_t)4096 * 64 * 4);
  float* X3 = (float*)carve((size_t)512 * 128 * 4);
  float* X4 = (float*)carve((size_t)64 * 256 * 4);
  float* D3 = (float*)carve((size_t)512 * 128 * 4);
  float* D2 = (float*)carve((size_t)4096 * 64 * 4);

  // gconv: gather -> pack weight -> WMMA gemm (raw conv output into outH)
  auto gconv = [&](const float* x, int d, int C, int nt, int wleaf, int Cout,
                   float* outH) {
    const int M = kN[d];
    const int Kreal = 7 * (C + nt);
    const int Kp = pad32(Kreal);
    const size_t ldsB = (size_t)8 * Kp * 4;
    gather_agg_kernel<<<M / 8, 256, ldsB, stream>>>(
        x, col[d], ety[d], nty[d], C, nt, Kp, agg, M);
    const int totalPack = (Kp / 32) * (Cout / 16) * 512;
    pack_w_kernel<<<(totalPack + 255) / 256, 256, 0, stream>>>(
        leaf(wleaf), Kreal, Kp, Cout, wpack);
    dim3 grid(M / 64, Cout / 16);
    gemm_bf16_kernel<<<grid, 128, 0, stream>>>(agg, wpack, outH, M, Kp, Cout);
  };

  // dense 1x1: pool/convert -> pack -> WMMA gemm (C is always mult of 32)
  auto dense = [&](const float* x, int nIn, int C, int P, int wleaf, int Cout,
                   float* outH) {
    const int nOut = nIn / P;
    const int total = nOut * C;
    pool_bf16_kernel<<<(total + 255) / 256, 256, 0, stream>>>(x, pooled, nOut, C, P);
    const int totalPack = (C / 32) * (Cout / 16) * 512;
    pack_w_kernel<<<(totalPack + 255) / 256, 256, 0, stream>>>(
        leaf(wleaf), C, C, Cout, wpack);
    dim3 grid(nOut / 64, Cout / 16);
    gemm_bf16_kernel<<<grid, 128, 0, stream>>>(pooled, wpack, outH, nOut, C, Cout);
  };

  // GroupNorm + activation (bleaf = index of 'b' leaf; 'g' = bleaf+1)
  auto gnact = [&](const float* h, int bleaf, const float* resPre,
                   const float* addPost, float* out, int nOut, int C, int shift) {
    const int grp = (C / 4 < 32) ? (C / 4) : 32;
    const int total = nOut * grp;
    gn_act_kernel<<<(total + 255) / 256, 256, 0, stream>>>(
        h, leaf(bleaf + 1), leaf(bleaf), resPre, addPost, out, nOut, C, grp, shift);
  };

  // Residual block: gelu(gn(conv2(gelu(gn(conv1(x))))) + x), in place allowed.
  auto resblk = [&](float* X, int d, int C, int nt, int base, float* outX) {
    gconv(X, d, C, nt, base + 2, C, bufH);
    gnact(bufH, base, nullptr, nullptr, bufT, kN[d], C, 0);
    gconv(bufT, d, C, nt, base + 5, C, bufH);
    gnact(bufH, base + 3, X, nullptr, outX, kN[d], C, 0);
  };

  // ---------------- forward ----------------
  // conv1 (6 -> 32) at depth 0
  gconv(data, 0, 6, 7, 2, 32, bufH);
  gnact(bufH, 0, nullptr, nullptr, bufX, kN[0], 32, 0);
  // enc0 (32 -> 32)
  gconv(bufX, 0, 32, 7, 11, 32, bufH);
  gnact(bufH, 9, nullptr, nullptr, bufT, kN[0], 32, 0);
  // down0 (32 -> 64), depth 0 -> 1
  dense(bufT, kN[0], 32, 8, 5, 64, bufH);
  gnact(bufH, 3, nullptr, nullptr, bufX, kN[1], 64, 0);
  // enc1 (64 -> 64)
  gconv(bufX, 1, 64, 6, 14, 64, bufH);
  gnact(bufH, 12, nullptr, nullptr, bufT, kN[1], 64, 0);
  // down1 (64 -> 64), depth 1 -> 2
  dense(bufT, kN[1], 64, 8, 8, 64, bufH);
  gnact(bufH, 6, nullptr, nullptr, X2, kN[2], 64, 0);

  // U-Net encoder
  resblk(X2, 2, 64, 5, 57, X2);   // net_enc0_0
  resblk(X2, 2, 64, 5, 63, X2);   // net_enc0_1   (X2 becomes skip[2])
  dense(X2, kN[2], 64, 8, 53, 128, bufH);        // net_down0
  gnact(bufH, 51, nullptr, nullptr, X3, kN[3], 128, 0);
  resblk(X3, 3, 128, 4, 69, X3);  // net_enc1_0
  resblk(X3, 3, 128, 4, 75, X3);  // net_enc1_1   (X3 becomes skip[3])
  dense(X3, kN[3], 128, 8, 56, 256, bufH);       // net_down1
  gnact(bufH, 54, nullptr, nullptr, X4, kN[4], 256, 0);
  resblk(X4, 4, 256, 3, 81, X4);  // net_enc2_0
  resblk(X4, 4, 256, 3, 87, X4);  // net_enc2_1

  // U-Net decoder
  resblk(X4, 4, 256, 3, 15, X4);  // net_dec0_0
  resblk(X4, 4, 256, 3, 21, X4);  // net_dec0_1
  dense(X4, kN[4], 256, 1, 95, 128, bufH);       // net_up0 (64x128)
  gnact(bufH, 93, nullptr, X3, D3, kN[3], 128, 3);  // repeat8 + skip[3]
  resblk(D3, 3, 128, 4, 27, D3);  // net_dec1_0
  resblk(D3, 3, 128, 4, 33, D3);  // net_dec1_1
  dense(D3, kN[3], 128, 1, 98, 64, bufH);        // net_up1 (512x64)
  gnact(bufH, 96, nullptr, X2, D2, kN[2], 64, 3);   // repeat8 + skip[2]
  resblk(D2, 2, 64, 5, 39, D2);   // net_dec2_0
  resblk(D2, 2, 64, 5, 45, (float*)d_out);  // net_dec2_1 -> output (4096x64)
}
